// NNBase_68848325754960
// MI455X (gfx1250) — matrix operand order repaired and checked by reading, not compiled, not run on hardware
//
#include <hip/hip_runtime.h>
#include <hip/hip_bf16.h>

// GRU scan for MI455X (gfx1250), wave32 + v_wmma_f32_16x16x32_bf16.
//
// T=128, N=1024, H=512, I=64, 3H=1536.
//   * Batch-tiled recurrence: each WG owns 32 batch rows for all 128 steps
//     (rows independent => zero cross-WG sync). 32 WGs.
//   * Two 16-row M-tiles per WG: every B tile fetched from L2 feeds 2 WMMAs
//     (halves L2 weight traffic; 2 independent WMMA chains hide XDL latency).
//   * Weights converted once to bf16 in d_ws (~1.7MB, L2-resident); all
//     streaming output uses non-temporal stores so it cannot evict them.
//   * Per step, gate GEMM runs in 2 column phases (c<256 / c>=256) so the
//     gate scratch fits in the 320KB WGP LDS (total ~238KB incl. bias LUT).
//   * gi projection (K=64) fused into the step; fp32 carried hidden state;
//     n-gate keeps gi_n / gh_n separate for tanh(i_n + r*h_n).

typedef __bf16 bf16_t;
typedef __attribute__((ext_vector_type(16))) __bf16 v16bf;
typedef __attribute__((ext_vector_type(8)))  __bf16 v8bf;
typedef __attribute__((ext_vector_type(8)))  float  v8f;

#define TT 128
#define NN 1024
#define HH 512
#define II 64
#define G3 1536
#define BM 32          // batch rows per workgroup (2 M-tiles)
#define TPB 256        // 8 waves (wave32)
#define TPW 6          // tiles per wave per phase (48 tiles / 8 waves)
// padded LDS strides (16B alignment for v8bf loads; rows +8 apart land on
// bank offset +32 so the two lane-halves of a wave never collide)
#define AHS 520
#define AXS 72
#define GS2 772        // 768 gate cols (r|z|hn for one phase) + pad
#define GNS 260        // 256 gi_n cols + pad

__global__ void cvt_weights(const float* __restrict__ Wih,
                            const float* __restrict__ Whh,
                            bf16_t* __restrict__ WihB,
                            bf16_t* __restrict__ WhhB) {
  const int stride = gridDim.x * blockDim.x;
  const int i0 = blockIdx.x * blockDim.x + threadIdx.x;
  for (int k = i0; k < G3 * HH; k += stride) WhhB[k] = (bf16_t)Whh[k];
  for (int k = i0; k < G3 * II; k += stride) WihB[k] = (bf16_t)Wih[k];
}

__launch_bounds__(TPB, 1)
__global__ void gru_scan(const float* __restrict__ hxs,
                         const float* __restrict__ masks,
                         const float* __restrict__ x_all,   // [T,N,I]
                         const bf16_t* __restrict__ WihB,   // [3H,I] bf16
                         const bf16_t* __restrict__ WhhB,   // [3H,H] bf16
                         const float* __restrict__ b_ih,
                         const float* __restrict__ b_hh,
                         float* __restrict__ out) {
  __shared__ float  h_cur[BM * HH];   // fp32 hidden state (carried)   64.0KB
  __shared__ bf16_t a_h[BM * AHS];    // masked h snapshot, bf16 A     33.3KB
  __shared__ bf16_t a_x[BM * AXS];    // masked x, bf16 A               4.6KB
  __shared__ float  g[BM * GS2];      // phase gates: r|z|gh_n         98.8KB
  __shared__ float  g_in[BM * GNS];   // phase gi_n                    33.3KB
  __shared__ float  bias[4 * HH];     // r|z combined, i_n, h_n         8.0KB
  __shared__ float  msk[BM];

  const int tid     = threadIdx.x;
  const int rowbase = blockIdx.x * BM;
  const int lane    = tid & 31;
  const int wave    = tid >> 5;
  const int lr      = lane & 15;      // A: M row / B,D: column within tile
  const int hi      = lane >> 4;      // K / M half selector

  for (int i = tid; i < BM * HH; i += TPB) {
    int r = i >> 9, c = i & 511;
    h_cur[i] = hxs[(rowbase + r) * HH + c];
  }
  // Bias LUT: biases are read 6x per element per step otherwise.
  for (int c = tid; c < HH; c += TPB) {
    bias[c]          = b_ih[c] + b_hh[c];                  // r pre-act
    bias[HH + c]     = b_ih[HH + c] + b_hh[HH + c];        // z pre-act
    bias[2 * HH + c] = b_ih[2 * HH + c];                   // i_n
    bias[3 * HH + c] = b_hh[2 * HH + c];                   // h_n (inside r*)
  }
  __syncthreads();

  for (int t = 0; t < TT; ++t) {
    if (tid < BM) msk[tid] = masks[t * NN + rowbase + tid];
    __syncthreads();

    // Frozen masked bf16 A operands for this step.
    for (int i = tid; i < BM * HH; i += TPB) {
      int r = i >> 9, c = i & 511;
      a_h[r * AHS + c] = (bf16_t)(h_cur[r * HH + c] * msk[r]);
    }
    for (int i = tid; i < BM * II; i += TPB) {
      int r = i >> 6, c = i & 63;
      float xv = __builtin_nontemporal_load(
          &x_all[((size_t)t * NN + rowbase + r) * II + c]);
      a_x[r * AXS + c] = (bf16_t)(xv * msk[r]);
    }
    __syncthreads();

    for (int p = 0; p < 2; ++p) {  // column phase: c in [p*256, p*256+256)
      // ---- GEMM: 48 tiles (r,z,n gates restricted to this phase's cols) ---
      for (int j = 0; j < TPW; ++j) {
        const int q    = wave * TPW + j;
        const int gate = q >> 4;                           // 0=r 1=z 2=n
        const int cidx = q & 15;
        const int cb_g = gate * HH + p * 256 + cidx * 16;  // global gate col
        const int cb_s = gate * 256 + cidx * 16;           // LDS gate col
        v8f ax0 = {}, ax1 = {}, ah0 = {}, ah1 = {};
        { // x @ W_ih^T : K = 64
          const bf16_t* brow = WihB + (size_t)(cb_g + lr) * II;
#pragma unroll
          for (int ki = 0; ki < 2; ++ki) {
            const int kb = ki * 32;
            v16bf b   = *(const v16bf*)(brow + kb + hi * 16);
            v8bf al0  = *(const v8bf*)(&a_x[lr * AXS + kb + hi * 16]);
            v8bf ah0v = *(const v8bf*)(&a_x[lr * AXS + kb + hi * 16 + 8]);
            v8bf al1  = *(const v8bf*)(&a_x[(16 + lr) * AXS + kb + hi * 16]);
            v8bf ah1v = *(const v8bf*)(&a_x[(16 + lr) * AXS + kb + hi * 16 + 8]);
            v16bf A0 = __builtin_shufflevector(al0, ah0v, 0, 1, 2, 3, 4, 5, 6,
                                               7, 8, 9, 10, 11, 12, 13, 14, 15);
            v16bf A1 = __builtin_shufflevector(al1, ah1v, 0, 1, 2, 3, 4, 5, 6,
                                               7, 8, 9, 10, 11, 12, 13, 14, 15);
            ax0 = __builtin_amdgcn_wmma_f32_16x16x32_bf16(
                false, A0, false, b, (short)0, ax0, false, false);
            ax1 = __builtin_amdgcn_wmma_f32_16x16x32_bf16(
                false, A1, false, b, (short)0, ax1, false, false);
          }
        }
        { // h @ W_hh^T : K = 512 (one B load feeds both M-tiles)
          const bf16_t* brow = WhhB + (size_t)(cb_g + lr) * HH;
#pragma unroll 4
          for (int ki = 0; ki < 16; ++ki) {
            const int kb = ki * 32;
            v16bf b   = *(const v16bf*)(brow + kb + hi * 16);
            v8bf al0  = *(const v8bf*)(&a_h[lr * AHS + kb + hi * 16]);
            v8bf ah0v = *(const v8bf*)(&a_h[lr * AHS + kb + hi * 16 + 8]);
            v8bf al1  = *(const v8bf*)(&a_h[(16 + lr) * AHS + kb + hi * 16]);
            v8bf ah1v = *(const v8bf*)(&a_h[(16 + lr) * AHS + kb + hi * 16 + 8]);
            v16bf A0 = __builtin_shufflevector(al0, ah0v, 0, 1, 2, 3, 4, 5, 6,
                                               7, 8, 9, 10, 11, 12, 13, 14, 15);
            v16bf A1 = __builtin_shufflevector(al1, ah1v, 0, 1, 2, 3, 4, 5, 6,
                                               7, 8, 9, 10, 11, 12, 13, 14, 15);
            ah0 = __builtin_amdgcn_wmma_f32_16x16x32_bf16(
                false, A0, false, b, (short)0, ah0, false, false);
            ah1 = __builtin_amdgcn_wmma_f32_16x16x32_bf16(
                false, A1, false, b, (short)0, ah1, false, false);
          }
        }
        // D layout: element v -> row hi*8+v (+16 for M-tile 1), column lr.
        if (gate < 2) {  // r,z: gi and gh add
#pragma unroll
          for (int v = 0; v < 8; ++v) {
            g[(hi * 8 + v) * GS2 + cb_s + lr]        = ah0[v] + ax0[v];
            g[(16 + hi * 8 + v) * GS2 + cb_s + lr]   = ah1[v] + ax1[v];
          }
        } else {         // n: keep gi_n and gh_n separate
          const int cn = cidx * 16;
#pragma unroll
          for (int v = 0; v < 8; ++v) {
            g[(hi * 8 + v) * GS2 + cb_s + lr]        = ah0[v];
            g[(16 + hi * 8 + v) * GS2 + cb_s + lr]   = ah1[v];
            g_in[(hi * 8 + v) * GNS + cn + lr]       = ax0[v];
            g_in[(16 + hi * 8 + v) * GNS + cn + lr]  = ax1[v];
          }
        }
      }
      __syncthreads();

      // ---- Gate epilogue + state update for this phase's columns ----
      for (int i = tid; i < BM * 256; i += TPB) {
        int r = i >> 8, cl = i & 255;
        int c = p * 256 + cl;
        float m      = msk[r];
        float pre_r  = g[r * GS2 + cl] + bias[c];
        float pre_z  = g[r * GS2 + 256 + cl] + bias[HH + c];
        float rg     = 1.0f / (1.0f + __expf(-pre_r));
        float zg     = 1.0f / (1.0f + __expf(-pre_z));
        float hn_pre = g[r * GS2 + 512 + cl] + bias[3 * HH + c];  // h_n
        float in_pre = g_in[r * GNS + cl] + bias[2 * HH + c];     // i_n
        float ng     = tanhf(in_pre + rg * hn_pre);
        float hm     = h_cur[r * HH + c] * m;                     // cell uses h*m
        float hnew   = (1.0f - zg) * ng + zg * hm;
        h_cur[r * HH + c] = hnew;                                 // own-element RMW
        // streaming output: NT store keeps the bf16 weights L2-resident
        __builtin_nontemporal_store(
            hnew, &out[((size_t)t * NN + rowbase + r) * HH + c]);
      }
      __syncthreads();
    }
  }

  // h_final
  for (int i = tid; i < BM * HH; i += TPB) {
    int r = i >> 9, c = i & 511;
    __builtin_nontemporal_store(
        h_cur[r * HH + c],
        &out[(size_t)TT * NN * HH + (size_t)(rowbase + r) * HH + c]);
  }
}

extern "C" void kernel_launch(void* const* d_in, const int* in_sizes, int n_in,
                              void* d_out, int out_size, void* d_ws,
                              size_t ws_size, hipStream_t stream) {
  (void)in_sizes; (void)n_in; (void)out_size; (void)ws_size;
  const float* hxs   = (const float*)d_in[0];
  // d_in[1] (gru_init) is dead code in the reference.
  const float* masks = (const float*)d_in[2];
  const float* x_all = (const float*)d_in[3];
  const float* Wih   = (const float*)d_in[4];
  const float* Whh   = (const float*)d_in[5];
  const float* b_ih  = (const float*)d_in[6];
  const float* b_hh  = (const float*)d_in[7];

  // Workspace: bf16 weight copies (needs ~1.73 MB).
  bf16_t* WhhB = (bf16_t*)d_ws;
  bf16_t* WihB = WhhB + (size_t)G3 * HH;

  cvt_weights<<<512, 256, 0, stream>>>(Wih, Whh, WihB, WhhB);
  gru_scan<<<NN / BM, TPB, 0, stream>>>(hxs, masks, x_all, WihB, WhhB, b_ih,
                                        b_hh, (float*)d_out);
}
